// MTREbbinghausLayer_85289460563991
// MI455X (gfx1250) — compile-verified, matrix operand order checked by reading, not run on hardware
//
#include <hip/hip_runtime.h>
#include <hip/hip_bf16.h>
#include <math.h>

typedef __attribute__((ext_vector_type(16))) _Float16 v16h;
typedef __attribute__((ext_vector_type(8)))  float    v8f;

#define D_DIM 1024
#define N_DIM 256
#define SQN   16
#define B_DIM 8
#define S_DIM 1024
#define ROWS  (B_DIM * S_DIM)   // 8192
#define LR_C         0.1f
#define BASE_DECAY_C 0.05f

// ---------------------------------------------------------------------------
// f32 memory -> f16 WMMA A/B fragment (16x32 16-bit layout).
// lane 0-15: M=lane, K={0..7,16..23}; lane 16-31: M=lane-16, K={8..15,24..31}.
// p must already include the (lane>=16 ? +8 : 0) K offset.
// ---------------------------------------------------------------------------
__device__ __forceinline__ v16h cvt_frag(const float* __restrict__ p) {
    v16h f;
#pragma unroll
    for (int e = 0; e < 8; ++e) f[e]     = (_Float16)p[e];
#pragma unroll
    for (int e = 0; e < 8; ++e) f[8 + e] = (_Float16)p[16 + e];
    return f;
}

// ---------------------------------------------------------------------------
// C1 = A(MxK) @ B1(NxK)^T (+bias); if DUAL also C2 = A @ B2^T with the same
// A fragments. Each wave computes a 16 x (16*NT) strip: one A fragment feeds
// NT (x2 if DUAL) WMMAs per K-chunk.
// ---------------------------------------------------------------------------
template <int NT, bool DUAL>
__global__ void wmma_gemm_xt(const float* __restrict__ A, int Kdim,
                             const float* __restrict__ B1,
                             const float* __restrict__ B2,
                             const float* __restrict__ bias,
                             float* __restrict__ C1,
                             float* __restrict__ C2,
                             int M, int N) {
    const int lane = threadIdx.x & 31;
    const int wave = threadIdx.x >> 5;
    const int tile = blockIdx.x * 8 + wave;
    const int ntn  = N / (16 * NT);
    const int m_tile = tile / ntn;
    const int n_base = (tile - m_tile * ntn) * (16 * NT);
    if (m_tile * 16 >= M) return;               // wave-uniform

    const int mrow = m_tile * 16 + (lane & 15);
    const int koff = (lane >= 16) ? 8 : 0;

    const float* pa = A + (size_t)mrow * Kdim + koff;
    const float* pb1[NT];
    const float* pb2[NT];
#pragma unroll
    for (int nt = 0; nt < NT; ++nt) {
        const int nrow = n_base + nt * 16 + (lane & 15);
        pb1[nt] = B1 + (size_t)nrow * Kdim + koff;
        pb2[nt] = DUAL ? (B2 + (size_t)nrow * Kdim + koff) : nullptr;
    }

    v8f zero = {};
    v8f acc1[NT], acc2[NT];
#pragma unroll
    for (int nt = 0; nt < NT; ++nt) { acc1[nt] = zero; acc2[nt] = zero; }

    for (int kk = 0; kk < Kdim; kk += 32) {
        const v16h a = cvt_frag(pa + kk);
#pragma unroll
        for (int nt = 0; nt < NT; ++nt) {
            const v16h b1 = cvt_frag(pb1[nt] + kk);
            acc1[nt] = __builtin_amdgcn_wmma_f32_16x16x32_f16(
                false, a, false, b1, (short)0, acc1[nt], false, false);
            if (DUAL) {
                const v16h b2 = cvt_frag(pb2[nt] + kk);
                acc2[nt] = __builtin_amdgcn_wmma_f32_16x16x32_f16(
                    false, a, false, b2, (short)0, acc2[nt], false, false);
            }
        }
    }

#pragma unroll
    for (int nt = 0; nt < NT; ++nt) {
        const int n  = n_base + nt * 16 + (lane & 15);
        const float bv = bias ? bias[n] : 0.0f;
#pragma unroll
        for (int r = 0; r < 8; ++r) {
            const int m = m_tile * 16 + r + ((lane >= 16) ? 8 : 0);
            C1[(size_t)m * N + n] = acc1[nt][r] + bv;
            if (DUAL) C2[(size_t)m * N + n] = acc2[nt][r];
        }
    }
}

// ---------------------------------------------------------------------------
// gate = sigmoid(x @ Wg^T + bg): one wave per row, wave32 shuffle reduction.
// ---------------------------------------------------------------------------
__global__ void gate_kernel(const float* __restrict__ x,
                            const float* __restrict__ Wg,
                            const float* __restrict__ bg,
                            float* __restrict__ gate, int rows, int D) {
    const int lane = threadIdx.x & 31;
    const int row  = (blockIdx.x * blockDim.x + threadIdx.x) >> 5;
    if (row >= rows) return;
    const float* xr = x + (size_t)row * D;
    float s = 0.0f;
    for (int j = lane; j < D; j += 32) s += xr[j] * Wg[j];
#pragma unroll
    for (int off = 16; off > 0; off >>= 1) s += __shfl_xor(s, off, 32);
    if (lane == 0) gate[row] = 1.0f / (1.0f + expf(-(s + bg[0])));
}

// ---------------------------------------------------------------------------
// CDNA5 async global->LDS copy (ASYNCcnt-tracked). The generic address of a
// __shared__ object carries the LDS byte offset in its low 32 bits.
// ---------------------------------------------------------------------------
__device__ __forceinline__ void async_copy_f32_to_lds(float* lds_dst,
                                                      const float* gsrc) {
    unsigned lds_off = (unsigned)(size_t)lds_dst;
    asm volatile("global_load_async_to_lds_b32 %0, %1, off"
                 :: "v"(lds_off), "v"(gsrc)
                 : "memory");
}
__device__ __forceinline__ void wait_asynccnt0() {
    asm volatile("s_wait_asynccnt 0x0" ::: "memory");
}

// ---------------------------------------------------------------------------
// Fast-weight scan. One 256-thread workgroup per batch; W1/W2 (2 x 16KB) live
// in LDS for all 1024 steps. kt/vt for step t+1 stream asynchronously into a
// double buffer (global_load_async_to_lds_b32) while step t's VALU work runs;
// the step-boundary s_wait_asynccnt + barrier publishes them.
// Thread (i = tid>>4, k = tid&15) owns (head i, col k) of each vector and row
// p = k of each per-head 16x16 weight slab.
// ---------------------------------------------------------------------------
__global__ void scan_kernel(const float* __restrict__ Kp,
                            const float* __restrict__ Vp,
                            const float* __restrict__ gate,
                            float* __restrict__ Y,
                            float* __restrict__ err_out, int S) {
    __shared__ float w1[SQN * 256];
    __shared__ float w2[SQN * 256];
    __shared__ float ktb[2][256], vtb[2][256];
    __shared__ float hid[256], err[256], gh[256];
    __shared__ float activity[16], strength[16], last_seen[16], decay[16];
    __shared__ float s_part[16];
    __shared__ float s_errnorm, s_gt;

    const int tid = threadIdx.x;
    const int b   = blockIdx.x;
    const int i   = tid >> 4;
    const int k   = tid & 15;

    // init state: W1 = W2 = identity per head; strength = last_seen = 0
#pragma unroll
    for (int q = 0; q < 16; ++q) {
        const int e  = tid * 16 + q;
        const float v = (((e >> 4) & 15) == (e & 15)) ? 1.0f : 0.0f;
        w1[e] = v;
        w2[e] = v;
    }
    if (tid < 16) { strength[tid] = 0.0f; last_seen[tid] = 0.0f; }

    // prologue: async-stage step 0 operands
    {
        const size_t r0 = (size_t)b * S * 256;
        async_copy_f32_to_lds(&ktb[0][tid], Kp + r0 + tid);
        async_copy_f32_to_lds(&vtb[0][tid], Vp + r0 + tid);
        wait_asynccnt0();
    }
    __syncthreads();

    for (int t = 0; t < S; ++t) {
        const int cur = t & 1;
        const int nxt = cur ^ 1;
        const size_t rbase = ((size_t)b * S + t) * 256;

        // prefetch step t+1 operands into the other buffer (in flight during
        // this step's compute; drained at the end-of-step barrier)
        if (t + 1 < S) {
            async_copy_f32_to_lds(&ktb[nxt][tid], Kp + rbase + 256 + tid);
            async_copy_f32_to_lds(&vtb[nxt][tid], Vp + rbase + 256 + tid);
        }
        if (tid == 0) s_gt = gate[(size_t)b * S + t];
        const float* kt = ktb[cur];
        const float* vt = vtb[cur];

        // hidden[i][k] = sum_j kt[i][j] * w1[i][j][k]
        float h = 0.0f;
#pragma unroll
        for (int j = 0; j < 16; ++j) h += kt[i * 16 + j] * w1[i * 256 + j * 16 + k];
        hid[tid] = h;
        __syncthreads();

        // vt_pred[i][k] = sum_j hidden[j][i] * w2[i][j][k]   (hr = hidden^T)
        float vp = 0.0f;
#pragma unroll
        for (int j = 0; j < 16; ++j) vp += hid[j * 16 + i] * w2[i * 256 + j * 16 + k];
        const float e = vp - vt[tid];
        err[tid] = e;
        __syncthreads();

        // per-head reductions + global error norm
        if (tid < 16) {
            float sa = 0.0f, se = 0.0f;
#pragma unroll
            for (int j = 0; j < 16; ++j) {
                const float kv = kt[tid * 16 + j]; sa += kv * kv;
                const float ev = err[tid * 16 + j]; se += ev * ev;
            }
            activity[tid] = sqrtf(sa);
            s_part[tid]   = se;
        }
        __syncthreads();
        if (tid == 0) {
            float se = 0.0f;
#pragma unroll
            for (int j = 0; j < 16; ++j) se += s_part[j];
            s_errnorm = sqrtf(se);
        }
        __syncthreads();

        const float tf = (float)(t + 1);
        if (tid < 16) {
            const float dt = tf - last_seen[tid];
            const float tg = 1.0f / fmaxf(log1pf(dt * 0.1f), 1.0f);
            decay[tid] = BASE_DECAY_C / (1.0f + strength[tid] * tg);   // old strength
            const float rq      = 1.0f / (1.0f + s_errnorm);
            const float spacing = log1pf(dt) * activity[tid];
            strength[tid] += rq * spacing * s_gt * 0.05f;
            if (activity[tid] > 0.1f) last_seen[tid] = tf;
        }

        // grad_h[i][k] = sum_j err[i][j] * w2[i][k][j]  (contiguous w2 row)
        float g = 0.0f;
#pragma unroll
        for (int j = 0; j < 16; ++j) g += err[i * 16 + j] * w2[i * 256 + k * 16 + j];
        gh[tid] = g;
        __syncthreads();   // all reads of w2 / decay done before updates

        // weight update + decay toward identity; thread owns (head i, row p=k)
        {
            const int p = k;
            const float lrg = LR_C * s_gt;
            const float df  = decay[i];
            const float ktp = kt[i * 16 + p];
            const float hrp = hid[p * 16 + i];        // hr[i][p]
#pragma unroll
            for (int q = 0; q < 16; ++q) {
                const float idq = (p == q) ? 1.0f : 0.0f;
                const int   e1  = i * 256 + p * 16 + q;
                const float nw1 = w1[e1] - lrg * ktp * gh[i * 16 + q];
                w1[e1] = nw1 * (1.0f - df) + idq * df;
                const float nw2 = w2[e1] - lrg * hrp * err[i * 16 + q];
                w2[e1] = nw2 * (1.0f - df) + idq * df;
            }
        }

        // yt[a*16+c] = vt_pred[c][a] -> thread (i,k) writes index k*16+i
        Y[rbase + k * 16 + i] = vp;
        if (tid == 0) err_out[(size_t)b * S + t] = s_errnorm;

        wait_asynccnt0();   // next step's kt/vt landed in LDS
        __syncthreads();    // publish them; protect kt/vt/hid/err reuse
    }
}

extern "C" void kernel_launch(void* const* d_in, const int* in_sizes, int n_in,
                              void* d_out, int out_size, void* d_ws, size_t ws_size,
                              hipStream_t stream) {
    (void)in_sizes; (void)n_in; (void)out_size; (void)ws_size;
    const float* x  = (const float*)d_in[0];
    const float* Wk = (const float*)d_in[1];
    const float* Wv = (const float*)d_in[2];
    const float* Wg = (const float*)d_in[3];
    const float* bg = (const float*)d_in[4];
    const float* Wo = (const float*)d_in[5];
    const float* bo = (const float*)d_in[6];

    float* out     = (float*)d_out;                     // (B,S,D) = 8388608 f32
    float* err_sig = out + (size_t)ROWS * D_DIM;        // (B,S,1) = 8192 f32

    float* ws   = (float*)d_ws;
    float* Kp   = ws;                                   // 8192*256
    float* Vp   = Kp + (size_t)ROWS * N_DIM;            // 8192*256
    float* gtv  = Vp + (size_t)ROWS * N_DIM;            // 8192
    float* Y    = gtv + ROWS;                           // 8192*256

    // Phase 1: fused K/V projections, 16x32 strip per wave (x read once)
    {
        const int waves = (ROWS / 16) * (N_DIM / 32);   // 4096
        wmma_gemm_xt<2, true><<<waves / 8, 256, 0, stream>>>(
            x, D_DIM, Wk, Wv, nullptr, Kp, Vp, ROWS, N_DIM);
        gate_kernel<<<(ROWS * 32 + 255) / 256, 256, 0, stream>>>(
            x, Wg, bg, gtv, ROWS, D_DIM);
    }

    // Phase 2: sequential fast-weight scan, one workgroup per batch
    scan_kernel<<<B_DIM, 256, 0, stream>>>(Kp, Vp, gtv, Y, err_sig, S_DIM);

    // Phase 3: output projection + bias, 16x64 strip per wave
    {
        const int waves = (ROWS / 16) * (D_DIM / 64);   // 8192
        wmma_gemm_xt<4, false><<<waves / 8, 256, 0, stream>>>(
            Y, N_DIM, Wo, nullptr, bo, out, nullptr, ROWS, D_DIM);
    }
}
